// HybridFeedForwardQuantum_65481071395578
// MI455X (gfx1250) — matrix operand order repaired and checked by reading, not compiled, not run on hardware
//
#include <hip/hip_runtime.h>

// ---------------------------------------------------------------------------
// Fused  out = relu( (cos(x)*cos(theta)) @ W1 + b1 ) @ W2 + b2
// x:[16384,768] f32, W1:[768,3072], W2:[3072,768], out:[16384,768] f32
// f16 WMMA (f32 accumulate); weights pre-transposed+converted to f16 in d_ws
// (L2-resident, 9.4MB). M-tile = 64 rows per workgroup (512 thr / 16 waves)
// to double B-fragment reuse vs 32-row tiling; h stays in LDS (double-
// buffered, one barrier per FFN chunk).
// ---------------------------------------------------------------------------

typedef __attribute__((ext_vector_type(16))) _Float16 v16h;
typedef __attribute__((ext_vector_type(8)))  _Float16 v8h;
typedef __attribute__((ext_vector_type(4)))  _Float16 v4h;
typedef __attribute__((ext_vector_type(8)))  float    v8f;

#define EMBED 768
#define FFN   3072
#define MTILE 64
#define THREADS 512               // 16 waves
#define FC    128                 // FFN chunk per iteration
#define QA_STRIDE 776             // 768 + 8 halfs pad (16B-chunk reads conflict-free)
#define HB_STRIDE 136             // 128 + 8 halfs pad

union V16U { v16h v; v8h h[2]; };

// A fragment (16x32 f16, ISA 7.12.2): lane = 16*khalf + m.
// VGPRs 0-7 = K[khalf*16 .. +16) (contiguous order chosen here, not the ISA table order; matches B) -> two b128 chunks
__device__ __forceinline__ v16h load_a_frag(const _Float16* p, int stride, int row,
                                            int k0, int khalf) {
  const _Float16* r = p + row * stride + k0;
  V16U a;
  a.h[0] = *(const v8h*)(r + khalf * 16);
  a.h[1] = *(const v8h*)(r + khalf * 16 + 8);
  return a.v;
}

__device__ __forceinline__ v8f wmma_f16(v16h a, v16h b, v8f c) {
  return __builtin_amdgcn_wmma_f32_16x16x32_f16(false, a, false, b, (short)0, c,
                                                false, false);
}

// --------------------- weight transpose + f32->f16 convert ------------------
// out[c*R + r] = (f16) in[r*C + c]
__global__ void __launch_bounds__(256)
transpose_cvt_f16(const float* __restrict__ in, _Float16* __restrict__ out,
                  int R, int C) {
  __shared__ float tile[32][33];
  int c  = blockIdx.x * 32 + threadIdx.x;
  int r0 = blockIdx.y * 32;
  for (int j = threadIdx.y; j < 32; j += 8)
    tile[j][threadIdx.x] = in[(size_t)(r0 + j) * C + c];
  __syncthreads();
  int r  = r0 + threadIdx.x;
  int cb = blockIdx.x * 32;
  for (int j = threadIdx.y; j < 32; j += 8)
    out[(size_t)(cb + j) * R + r] = (_Float16)tile[threadIdx.x][j];
}

// ------------------------------- fused FFN ----------------------------------
__global__ void __launch_bounds__(THREADS)
ffnq_kernel(const float* __restrict__ x, const float* __restrict__ theta,
            const float* __restrict__ b1, const float* __restrict__ b2,
            const _Float16* __restrict__ W1T,   // [FFN][EMBED]  (f-major)
            const _Float16* __restrict__ W2T,   // [EMBED][FFN]  (e-major)
            float* __restrict__ out) {
  __shared__ _Float16 qA[MTILE * QA_STRIDE];        // q tile, f16 row-major (~97KB)
  __shared__ _Float16 hB[2][MTILE * HB_STRIDE];     // h chunk, double-buffered
  __shared__ float    ctheta[EMBED];

  const int tid   = threadIdx.x;
  const int wave  = tid >> 5;          // 0..15
  const int lane  = tid & 31;
  const int khalf = lane >> 4;
  const int lid   = lane & 15;
  const int row0  = blockIdx.x * MTILE;

  // GEMM1 mapping: wave -> (n-tile, m-pair)
  const int nt1   = wave >> 1;         // 0..7  : 16-col tile within FC chunk
  const int mpair = wave & 1;          // 0..1  : rows [mpair*32, mpair*32+32)

  // ---- cos(theta) into LDS
  for (int i = tid; i < EMBED; i += THREADS) ctheta[i] = __cosf(theta[i]);
  __syncthreads();

  // ---- Phase A: q = cos(x)*cos(theta) -> f16 tile in LDS
  const float4* x4 = (const float4*)(x + (size_t)row0 * EMBED);
  for (int i = tid; i < MTILE * EMBED / 4; i += THREADS) {
    int e = i * 4;
    int r = e / EMBED;
    int c = e % EMBED;
    float4 xv = x4[i];
    float4 ct = *(const float4*)&ctheta[c];
    v4h qh;
    qh.x = (_Float16)(__cosf(xv.x) * ct.x);
    qh.y = (_Float16)(__cosf(xv.y) * ct.y);
    qh.z = (_Float16)(__cosf(xv.z) * ct.z);
    qh.w = (_Float16)(__cosf(xv.w) * ct.w);
    *(v4h*)&qA[r * QA_STRIDE + c] = qh;
  }
  __syncthreads();

  // ---- persistent GEMM2 accumulators: 3 n-tiles x 4 m-tiles per wave (96 VGPRs)
  v8f zero = {};
  v8f acc[12];
#pragma unroll
  for (int i = 0; i < 12; ++i) acc[i] = zero;

  // ---- loop over FFN chunks
  for (int ch = 0; ch < FFN / FC; ++ch) {
    const int c0 = ch * FC;
    _Float16* hcur = hB[ch & 1];

    // prefetch next chunk's W1 columns + this chunk's W2 rows
    if (ch + 1 < FFN / FC)
      __builtin_prefetch(&W1T[(size_t)(c0 + FC + nt1 * 16 + lid) * EMBED], 0, 1);
    __builtin_prefetch(&W2T[(size_t)(wave * 3 * 16 + lid) * FFN + c0], 0, 1);

    // ---- GEMM1: h_chunk[64 x 128] = q[64 x 768] @ W1[:, c0:c0+128]
    // wave handles n-tile nt1 and rows [mpair*32, mpair*32+32)
    const _Float16* bptr =
        &W1T[(size_t)(c0 + nt1 * 16 + lid) * EMBED + khalf * 16];
    v8f c0a = zero, c1a = zero;
#pragma unroll 4
    for (int ks = 0; ks < EMBED / 32; ++ks) {
      v16h a0 = load_a_frag(qA, QA_STRIDE, mpair * 32 + lid,      ks * 32, khalf);
      v16h a1 = load_a_frag(qA, QA_STRIDE, mpair * 32 + 16 + lid, ks * 32, khalf);
      v16h bf = *(const v16h*)(bptr + ks * 32);
      c0a = wmma_f16(a0, bf, c0a);
      c1a = wmma_f16(a1, bf, c1a);
    }

    float b1v = b1[c0 + nt1 * 16 + lid];

    // C layout: VGPR i -> (M = khalf*8 + i, N = lid)
#pragma unroll
    for (int i = 0; i < 8; ++i) {
      float h0 = c0a[i] + b1v; h0 = h0 > 0.f ? h0 : 0.f;
      float h1 = c1a[i] + b1v; h1 = h1 > 0.f ? h1 : 0.f;
      hcur[(mpair * 32 + khalf * 8 + i) * HB_STRIDE + nt1 * 16 + lid]      = (_Float16)h0;
      hcur[(mpair * 32 + 16 + khalf * 8 + i) * HB_STRIDE + nt1 * 16 + lid] = (_Float16)h1;
    }
    __syncthreads();   // hcur fully written; prior buffer reads already done per-wave

    // ---- GEMM2: acc += h_chunk[64 x 128] @ W2[c0:c0+128, :]
    // wave covers n-tiles [wave*3, wave*3+3), all 4 m-tiles
#pragma unroll
    for (int ks = 0; ks < FC / 32; ++ks) {
      v16h a[4];
#pragma unroll
      for (int mt = 0; mt < 4; ++mt)
        a[mt] = load_a_frag(hcur, HB_STRIDE, mt * 16 + lid, ks * 32, khalf);
#pragma unroll
      for (int nt = 0; nt < 3; ++nt) {
        int ecol = (wave * 3 + nt) * 16 + lid;
        v16h bf = *(const v16h*)&W2T[(size_t)ecol * FFN + c0 + ks * 32 + khalf * 16];
#pragma unroll
        for (int mt = 0; mt < 4; ++mt)
          acc[nt * 4 + mt] = wmma_f16(a[mt], bf, acc[nt * 4 + mt]);
      }
    }
  }

  // ---- epilogue: + b2, store f32 (coalesced 64B per half-wave)
  float* obase = out + (size_t)row0 * EMBED;
#pragma unroll
  for (int nt = 0; nt < 3; ++nt) {
    int ecol  = (wave * 3 + nt) * 16 + lid;
    float b2v = b2[ecol];
#pragma unroll
    for (int mt = 0; mt < 4; ++mt) {
#pragma unroll
      for (int i = 0; i < 8; ++i)
        obase[(size_t)(mt * 16 + khalf * 8 + i) * EMBED + ecol] =
            acc[nt * 4 + mt][i] + b2v;
    }
  }
}

// ---------------------------------------------------------------------------
extern "C" void kernel_launch(void* const* d_in, const int* in_sizes, int n_in,
                              void* d_out, int out_size, void* d_ws, size_t ws_size,
                              hipStream_t stream) {
  const float* x     = (const float*)d_in[0];
  const float* theta = (const float*)d_in[1];
  const float* W1    = (const float*)d_in[2];
  const float* b1    = (const float*)d_in[3];
  const float* W2    = (const float*)d_in[4];
  const float* b2    = (const float*)d_in[5];
  float* out = (float*)d_out;

  _Float16* W1T = (_Float16*)d_ws;                 // [FFN][EMBED] f16
  _Float16* W2T = W1T + (size_t)FFN * EMBED;       // [EMBED][FFN] f16

  dim3 tb(32, 8);
  // W1: [EMBED x FFN] -> W1T [FFN x EMBED]
  transpose_cvt_f16<<<dim3(FFN / 32, EMBED / 32), tb, 0, stream>>>(W1, W1T, EMBED, FFN);
  // W2: [FFN x EMBED] -> W2T [EMBED x FFN]
  transpose_cvt_f16<<<dim3(EMBED / 32, FFN / 32), tb, 0, stream>>>(W2, W2T, FFN, EMBED);

  int M = in_sizes[0] / EMBED;   // 16384 rows
  ffnq_kernel<<<M / MTILE, THREADS, 0, stream>>>(x, theta, b1, b2, W1T, W2T, out);
}
